// RGSACausalSelfAttention_50972671868993
// MI455X (gfx1250) — matrix operand order repaired and checked by reading, not compiled, not run on hardware
//
#include <hip/hip_runtime.h>
#include <math.h>

// ---------------------------------------------------------------------------
// Types & helpers
// ---------------------------------------------------------------------------
typedef __attribute__((ext_vector_type(16))) __bf16 v16bf;
typedef __attribute__((ext_vector_type(8)))  float  v8f;

typedef unsigned short u16;
typedef unsigned int   u32;

union Frag {               // one WMMA bf16 operand (8 VGPRs / lane)
    v16bf v;
    uint4 q[2];
    u32   u[8];
    u16   h[16];
};

__device__ __forceinline__ u16 f2bf(float f) {
    union { float f; u32 u; } x; x.f = f;
    u32 r = x.u + 0x7FFFu + ((x.u >> 16) & 1u);   // round-to-nearest-even
    return (u16)(r >> 16);
}

__device__ __forceinline__ v8f zero8() {
    v8f z;
#pragma unroll
    for (int i = 0; i < 8; ++i) z[i] = 0.0f;
    return z;
}

// ---------------------------------------------------------------------------
// Problem constants
// ---------------------------------------------------------------------------
#define BATCH   2
#define SEQ     2048
#define CDIM    1024
#define NHEAD   16
#define HDIM    64
#define NCHUNK  32
#define TOPB    8
#define ROWS    (BATCH * SEQ)        // 4096

// LDS strides (multiples of 4 dwords -> 16B-aligned b128 fragment loads)
#define ALD_STRIDE 40                // ushorts per A-tile row   (32 + 8 pad)
#define BGE_STRIDE 20                // dwords per B-tile column (16 + 4 pad)
#define ATT_STRIDE 36                // dwords per K/V-tile col  (32 + 4 pad)

// ---------------------------------------------------------------------------
// f32 -> bf16 row-major convert
// ---------------------------------------------------------------------------
__global__ void cvt_bf16_kernel(const float* __restrict__ in, u16* __restrict__ out, int n) {
    int i = blockIdx.x * blockDim.x + threadIdx.x;
    if (i < n) out[i] = f2bf(in[i]);
}

// ---------------------------------------------------------------------------
// Weight swizzle: f32 [K][N] -> fragment-major bf16 dwords [N][K/2]:
//   out[n*(K/2)+p] = {W[2p][n], W[2p+1][n]}
// A B-tile column is then 16 contiguous dwords -> straight uint4 LDS staging
// and 2x ds_load_b128 per WMMA fragment. Reads are coalesced; scattered
// writes are absorbed by the 192MB L2.
// ---------------------------------------------------------------------------
__global__ void swz_weight_kernel(const float* __restrict__ W, u32* __restrict__ out,
                                  int K, int N) {
    int id = blockIdx.x * blockDim.x + threadIdx.x;
    if (id >= (K / 2) * N) return;
    int p = id / N, n = id - p * N;
    u32 lo = f2bf(W[(size_t)(2 * p) * N + n]);
    u32 hi = f2bf(W[(size_t)(2 * p + 1) * N + n]);
    out[(size_t)n * (K / 2) + p] = lo | (hi << 16);
}

// ---------------------------------------------------------------------------
// Tiled WMMA GEMM, double-buffered LDS (one barrier per K-step, global loads
// for tile t+1 in flight under the WMMAs of tile t).
// A bf16 row-major [M,K]; Bswz fragment-major [N][K/2] dwords.
// Block = 256 threads (8 waves), tile 128(M) x 64(N), K-step 32.
// MODE 0: qkv epilogue (scatter q/k plain + v t-pair-major). MODE 1: f32 + bias.
// ---------------------------------------------------------------------------
template <int MODE>
__global__ __launch_bounds__(256) void gemm_wmma_kernel(
    const u16* __restrict__ A, const u32* __restrict__ Bswz,
    const float* __restrict__ bias, int M, int N, int K,
    float* __restrict__ fout, u16* __restrict__ q_s, u16* __restrict__ k_s,
    u16* __restrict__ v_swz) {
    __shared__ u16 Alds[2][128 * ALD_STRIDE];   // 2 x 10240 B
    __shared__ u32 Blds[2][64 * BGE_STRIDE];    // 2 x  5120 B

    const int tid  = threadIdx.x;
    const int wave = tid >> 5;
    const int lane = tid & 31;
    const int half = lane >> 4;
    const int lm   = lane & 15;
    const int wm   = wave >> 1;          // 0..3 -> 32-row slice
    const int wn   = wave & 1;           // 0..1 -> 32-col slice
    const int m0   = blockIdx.y * 128;
    const int n0   = blockIdx.x * 64;

    // staging coordinates
    const int rA   = tid >> 2;           // A rows rA and rA+64
    const int segA = tid & 3;            // 8-ushort segment in the 32-wide row
    const int colB = tid >> 2;           // B column (0..63)
    const int psB  = tid & 3;            // 4-dword segment of the 16-dword column

    const u16* aSrc0 = A + (size_t)(m0 + rA) * K + segA * 8;
    const u16* aSrc1 = aSrc0 + (size_t)64 * K;
    const u32* bSrc  = Bswz + (size_t)(n0 + colB) * (K / 2) + psB * 4;

    uint4 aR0, aR1, bR;
    auto ld = [&](int k0) {
        aR0 = *reinterpret_cast<const uint4*>(aSrc0 + k0);
        aR1 = *reinterpret_cast<const uint4*>(aSrc1 + k0);
        bR  = *reinterpret_cast<const uint4*>(bSrc + (k0 >> 1));
    };
    auto st = [&](int buf) {
        *reinterpret_cast<uint4*>(&Alds[buf][rA * ALD_STRIDE + segA * 8])        = aR0;
        *reinterpret_cast<uint4*>(&Alds[buf][(rA + 64) * ALD_STRIDE + segA * 8]) = aR1;
        *reinterpret_cast<uint4*>(&Blds[buf][colB * BGE_STRIDE + psB * 4])       = bR;
    };

    v8f acc[2][2];
#pragma unroll
    for (int i = 0; i < 2; ++i)
#pragma unroll
        for (int j = 0; j < 2; ++j) acc[i][j] = zero8();

    ld(0);
    st(0);
    __syncthreads();

    const int ntiles = K >> 5;
    for (int t = 0; t < ntiles; ++t) {
        const int buf = t & 1;
        if (t + 1 < ntiles) ld((t + 1) << 5);    // next tile's global loads in flight

        Frag af[2], bfg[2];
#pragma unroll
        for (int i = 0; i < 2; ++i) {
            const u16* base = &Alds[buf][(wm * 32 + i * 16 + lm) * ALD_STRIDE + half * 8];
            af[i].q[0] = *reinterpret_cast<const uint4*>(base);
            af[i].q[1] = *reinterpret_cast<const uint4*>(base + 16);
        }
#pragma unroll
        for (int j = 0; j < 2; ++j) {
            const u32* base = &Blds[buf][(wn * 32 + j * 16 + lm) * BGE_STRIDE + half * 4];
            bfg[j].q[0] = *reinterpret_cast<const uint4*>(base);
            bfg[j].q[1] = *reinterpret_cast<const uint4*>(base + 8);
        }
#pragma unroll
        for (int i = 0; i < 2; ++i)
#pragma unroll
            for (int j = 0; j < 2; ++j)
                acc[i][j] = __builtin_amdgcn_wmma_f32_16x16x32_bf16(
                    false, af[i].v, false, bfg[j].v, (short)0, acc[i][j], false, false);

        if (t + 1 < ntiles) {
            st(buf ^ 1);                          // fill the idle buffer
            __syncthreads();
        }
    }

    // --- epilogue ---
#pragma unroll
    for (int i = 0; i < 2; ++i) {
#pragma unroll
        for (int j = 0; j < 2; ++j) {
            int ncol = n0 + wn * 32 + j * 16 + lm;
            float bv = bias[ncol];
#pragma unroll
            for (int r = 0; r < 8; ++r) {
                int row = m0 + wm * 32 + i * 16 + r + 8 * half;
                float val = acc[i][j][r] + bv;
                if (MODE == 1) {
                    fout[(size_t)row * N + ncol] = val;
                } else {
                    int b = row >> 11, t = row & (SEQ - 1);
                    int which = ncol >> 10, rem = ncol & 1023;
                    int h = rem >> 6, d = rem & 63;
                    int bh = b * NHEAD + h;
                    u16 us = f2bf(val);
                    if (which == 0) {
                        q_s[((size_t)bh * SEQ + t) * HDIM + d] = us;
                    } else if (which == 1) {
                        k_s[((size_t)bh * SEQ + t) * HDIM + d] = us;
                    } else {
                        // v fragment-major: dword[(bh*HDIM + d)*(SEQ/2) + t/2]
                        size_t dw = ((size_t)bh * HDIM + d) * (SEQ / 2) + (t >> 1);
                        v_swz[dw * 2 + (t & 1)] = us;
                    }
                }
            }
        }
    }
}

// ---------------------------------------------------------------------------
// Fused causal flash attention (bf16 WMMA, in-place f32 online softmax).
// grid = (T/64, B*H), block = 128 (4 waves x 16 queries).
// K/V tiles staged fragment-major in LDS (2x ds_load_b128 per fragment);
// next tile prefetched with global_prefetch while the current one computes.
// ---------------------------------------------------------------------------
__global__ __launch_bounds__(128) void flash_attn_kernel(
    const u16* __restrict__ q_s, const u16* __restrict__ k_s,
    const u16* __restrict__ v_swz, u16* __restrict__ y_attn) {
    __shared__ u32 Klds[64 * ATT_STRIDE];   // [key][p=d/2]      9216 B
    __shared__ u32 Vlds[64 * ATT_STRIDE];   // [d][p=keypair]    9216 B
    __shared__ u16 Plds[4 * 16 * 64];       // per-wave P pane   8192 B

    const int tid  = threadIdx.x;
    const int w    = tid >> 5;
    const int lane = tid & 31;
    const int half = lane >> 4;
    const int lm   = lane & 15;
    const int bh   = blockIdx.y;
    const int b    = bh >> 4;
    const int h    = bh & 15;
    const int qb   = blockIdx.x;
    const int q0   = qb * 64;

    // k row-major bf16 [t][64] == dwords [t][32] (already d-pair packed)
    const u32* kplane = reinterpret_cast<const u32*>(k_s) + (size_t)bh * SEQ * (HDIM / 2);
    const u32* vplane = reinterpret_cast<const u32*>(v_swz) + (size_t)bh * HDIM * (SEQ / 2);

    // Q fragments straight from global (row-major bf16, A-operand layout)
    Frag qA[2];
    {
        const u16* qbase = q_s + ((size_t)bh * SEQ + q0 + w * 16 + lm) * HDIM;
#pragma unroll
        for (int ks = 0; ks < 2; ++ks) {
            qA[ks].q[0] = *reinterpret_cast<const uint4*>(qbase + ks * 32 + half * 8);
            qA[ks].q[1] = *reinterpret_cast<const uint4*>(qbase + ks * 32 + half * 8 + 16);
        }
    }

    v8f  o[4];
    float mrow[8], lrow[8];
#pragma unroll
    for (int t = 0; t < 4; ++t) o[t] = zero8();
#pragma unroll
    for (int r = 0; r < 8; ++r) { mrow[r] = -1e30f; lrow[r] = 0.0f; }

    for (int jb = 0; jb <= qb; ++jb) {
        // --- stage K/V tiles (column = 32 contiguous dwords + pad) ---
#pragma unroll
        for (int it = 0; it < 4; ++it) {
            int idx = tid + it * 128;
            int c = idx >> 3, ps = idx & 7;
            *reinterpret_cast<uint4*>(&Klds[c * ATT_STRIDE + ps * 4]) =
                *reinterpret_cast<const uint4*>(kplane + (size_t)(jb * 64 + c) * (HDIM / 2) + ps * 4);
            *reinterpret_cast<uint4*>(&Vlds[c * ATT_STRIDE + ps * 4]) =
                *reinterpret_cast<const uint4*>(vplane + (size_t)c * (SEQ / 2) + jb * 32 + ps * 4);
        }
        // --- prefetch next tile (K: contiguous 8KB; V: 64 strided 128B runs) ---
        if (jb < qb) {
            __builtin_prefetch(
                reinterpret_cast<const char*>(kplane + (size_t)(jb + 1) * 64 * (HDIM / 2)) +
                    tid * 64, 0, 3);
            __builtin_prefetch(
                vplane + (size_t)(tid >> 1) * (SEQ / 2) + (jb + 1) * 32 + (tid & 1) * 16, 0, 3);
        }
        __syncthreads();

        // --- S = Q K^T (4 key tiles x 2 d-slices) ---
        v8f s[4];
#pragma unroll
        for (int t = 0; t < 4; ++t) s[t] = zero8();
#pragma unroll
        for (int ks = 0; ks < 2; ++ks) {
#pragma unroll
            for (int t = 0; t < 4; ++t) {
                Frag kf;
                const u32* base = &Klds[(t * 16 + lm) * ATT_STRIDE + ks * 16 + half * 4];
                kf.q[0] = *reinterpret_cast<const uint4*>(base);
                kf.q[1] = *reinterpret_cast<const uint4*>(base + 8);
                s[t] = __builtin_amdgcn_wmma_f32_16x16x32_bf16(
                    false, qA[ks].v, false, kf.v, (short)0, s[t], false, false);
            }
        }

        // --- scale + causal mask, in place ---
        const bool diag = (jb == qb);
#pragma unroll
        for (int t = 0; t < 4; ++t) {
            int kcol = jb * 64 + t * 16 + lm;
#pragma unroll
            for (int r = 0; r < 8; ++r) {
                float sv = s[t][r] * 0.125f;        // 1/sqrt(64)
                if (diag && kcol > (q0 + w * 16 + r + 8 * half)) sv = -1e30f;
                s[t][r] = sv;
            }
        }

        // --- online softmax (row r+8*half shared by the 16 lanes of one half) ---
#pragma unroll
        for (int r = 0; r < 8; ++r) {
            float mnew = mrow[r];
#pragma unroll
            for (int t = 0; t < 4; ++t) mnew = fmaxf(mnew, s[t][r]);
#pragma unroll
            for (int msk = 8; msk >= 1; msk >>= 1) mnew = fmaxf(mnew, __shfl_xor(mnew, msk, 32));
            float alpha = __expf(mrow[r] - mnew);
            mrow[r] = mnew;
            float rsum = 0.0f;
#pragma unroll
            for (int t = 0; t < 4; ++t) {
                float p = __expf(s[t][r] - mnew);
                s[t][r] = p;
                rsum += p;
            }
#pragma unroll
            for (int msk = 8; msk >= 1; msk >>= 1) rsum += __shfl_xor(rsum, msk, 32);
            lrow[r] = lrow[r] * alpha + rsum;
#pragma unroll
            for (int t = 0; t < 4; ++t) o[t][r] *= alpha;
        }

        // --- transpose P through per-wave LDS pane into A-fragment layout ---
#pragma unroll
        for (int t = 0; t < 4; ++t)
#pragma unroll
            for (int r = 0; r < 8; ++r)
                Plds[w * 1024 + (r + 8 * half) * 64 + t * 16 + lm] = f2bf(s[t][r]);
        asm volatile("s_wait_dscnt 0" ::: "memory");   // wave-local LDS RAW fence

        Frag pf[2];
#pragma unroll
        for (int ks = 0; ks < 2; ++ks) {
            const u16* base = &Plds[w * 1024 + lm * 64 + ks * 32 + half * 8];
            pf[ks].q[0] = *reinterpret_cast<const uint4*>(base);
            pf[ks].q[1] = *reinterpret_cast<const uint4*>(base + 16);
        }

        // --- O += P V ---
#pragma unroll
        for (int ks = 0; ks < 2; ++ks) {
#pragma unroll
            for (int t = 0; t < 4; ++t) {
                Frag vf;
                const u32* base = &Vlds[(t * 16 + lm) * ATT_STRIDE + ks * 16 + half * 4];
                vf.q[0] = *reinterpret_cast<const uint4*>(base);
                vf.q[1] = *reinterpret_cast<const uint4*>(base + 8);
                o[t] = __builtin_amdgcn_wmma_f32_16x16x32_bf16(
                    false, pf[ks].v, false, vf.v, (short)0, o[t], false, false);
            }
        }
        __syncthreads();
    }

    // --- normalize + write y_attn bf16 [B,T,C] ---
#pragma unroll
    for (int r = 0; r < 8; ++r) {
        float inv = 1.0f / lrow[r];
        int trow = q0 + w * 16 + r + 8 * half;
#pragma unroll
        for (int t = 0; t < 4; ++t)
            y_attn[((size_t)(b * SEQ + trow)) * CDIM + h * HDIM + t * 16 + lm] =
                f2bf(o[t][r] * inv);
    }
}

// ---------------------------------------------------------------------------
// chunk means: grid = B*NCHUNK, block = 256 (4 cols per thread)
// ---------------------------------------------------------------------------
__global__ void chunk_mean_kernel(const float* __restrict__ x, float* __restrict__ cm) {
    int blk = blockIdx.x;               // b*NCHUNK + n
    int b = blk >> 5, n = blk & 31;
    int c0 = threadIdx.x * 4;
    float s0 = 0, s1 = 0, s2 = 0, s3 = 0;
    const float* base = x + ((size_t)b * SEQ + n * 64) * CDIM + c0;
    for (int r = 0; r < 64; ++r) {
        s0 += base[0]; s1 += base[1]; s2 += base[2]; s3 += base[3];
        base += CDIM;
    }
    float* out = cm + (size_t)blk * CDIM + c0;
    out[0] = s0 * (1.0f / 64.0f); out[1] = s1 * (1.0f / 64.0f);
    out[2] = s2 * (1.0f / 64.0f); out[3] = s3 * (1.0f / 64.0f);
}

// ---------------------------------------------------------------------------
// router embeds: grid = B*NCHUNK, block = 32. writes raw embeds + normalized copy
// ---------------------------------------------------------------------------
__global__ void router_kernel(const float* __restrict__ cm, const float* __restrict__ Wr,
                              const float* __restrict__ br, float* __restrict__ embeds_out,
                              float* __restrict__ re_n) {
    int row = blockIdx.x;
    int j = threadIdx.x;
    float acc = br[j];
    const float* c = cm + (size_t)row * CDIM;
    for (int k = 0; k < CDIM; ++k) acc += c[k] * Wr[(size_t)k * 32 + j];
    embeds_out[(size_t)row * 32 + j] = acc;
    float ss = acc * acc;
#pragma unroll
    for (int msk = 16; msk >= 1; msk >>= 1) ss += __shfl_xor(ss, msk, 32);
    float nrm = fmaxf(sqrtf(ss), 1e-12f);
    re_n[(size_t)row * 32 + j] = acc / nrm;
}

// ---------------------------------------------------------------------------
// q_rt = normalize(x @ Wq_rt + b): grid = B*T, block = 32
// ---------------------------------------------------------------------------
__global__ void qrt_kernel(const float* __restrict__ x, const float* __restrict__ Wq,
                           const float* __restrict__ bq, float* __restrict__ q_rt_n) {
    int row = blockIdx.x;
    int j = threadIdx.x;
    float acc = bq[j];
    const float* xr = x + (size_t)row * CDIM;
    for (int k = 0; k < CDIM; ++k) acc += xr[k] * Wq[(size_t)k * 32 + j];
    float ss = acc * acc;
#pragma unroll
    for (int msk = 16; msk >= 1; msk >>= 1) ss += __shfl_xor(ss, msk, 32);
    float nrm = fmaxf(sqrtf(ss), 1e-12f);
    q_rt_n[(size_t)row * 32 + j] = acc / nrm;
}

// ---------------------------------------------------------------------------
// routing scores + top-8 (sorted desc, ties -> lower index): grid = B*T, block = 32
// ---------------------------------------------------------------------------
__global__ void score_topk_kernel(const float* __restrict__ q_rt_n,
                                  const float* __restrict__ re_n,
                                  float* __restrict__ scores_out, int* __restrict__ idx_out) {
    int row = blockIdx.x;
    int b = row >> 11;
    int n = threadIdx.x;
    float sc = 0.0f;
    const float* qv = q_rt_n + (size_t)row * 32;
    const float* rv = re_n + ((size_t)b * 32 + n) * 32;
#pragma unroll
    for (int k = 0; k < 32; ++k) sc += qv[k] * rv[k];
    scores_out[(size_t)row * 32 + n] = sc;

    float mysc = sc;
#pragma unroll
    for (int k = 0; k < TOPB; ++k) {
        float v = mysc;
        int   i = n;
#pragma unroll
        for (int msk = 16; msk >= 1; msk >>= 1) {
            float ov = __shfl_xor(v, msk, 32);
            int   oi = __shfl_xor(i, msk, 32);
            if (ov > v || (ov == v && oi < i)) { v = ov; i = oi; }
        }
        if (n == 0) idx_out[(size_t)row * TOPB + k] = i;
        if (n == i) mysc = -1e30f;
    }
}

// ---------------------------------------------------------------------------
// kernel_launch
// ---------------------------------------------------------------------------
extern "C" void kernel_launch(void* const* d_in, const int* in_sizes, int n_in,
                              void* d_out, int out_size, void* d_ws, size_t ws_size,
                              hipStream_t stream) {
    const float* x    = (const float*)d_in[0];
    const float* Wqkv = (const float*)d_in[1];
    const float* bqkv = (const float*)d_in[2];
    const float* Wpr  = (const float*)d_in[3];
    const float* bpr  = (const float*)d_in[4];
    const float* Wro  = (const float*)d_in[5];
    const float* bro  = (const float*)d_in[6];
    const float* Wqr  = (const float*)d_in[7];
    const float* bqr  = (const float*)d_in[8];

    // output layout (flat f32 buffer): y | chunk_indices(int) | routing_scores | routing_embeds
    float* y_out      = (float*)d_out;
    int*   idx_out    = (int*)((float*)d_out + (size_t)ROWS * CDIM);
    float* scores_out = (float*)d_out + (size_t)ROWS * CDIM + (size_t)ROWS * TOPB;
    float* embeds_out = scores_out + (size_t)ROWS * 32;

    // workspace carve-up
    char* p = (char*)d_ws;
    auto take = [&](size_t bytes) { char* r = p; p += (bytes + 255) & ~size_t(255); return r; };
    u16* xb        = (u16*)take((size_t)ROWS * CDIM * 2);                 // 8 MB
    u32* wqkv_swz  = (u32*)take((size_t)(CDIM / 2) * (3 * CDIM) * 4);     // 6 MB
    u32* wproj_swz = (u32*)take((size_t)(CDIM / 2) * CDIM * 4);           // 2 MB
    u16* q_s       = (u16*)take((size_t)BATCH * NHEAD * SEQ * HDIM * 2);  // 8 MB
    u16* k_s       = (u16*)take((size_t)BATCH * NHEAD * SEQ * HDIM * 2);  // 8 MB
    u16* v_swz     = (u16*)take((size_t)BATCH * NHEAD * SEQ * HDIM * 2);  // 8 MB
    u16* y_attn    = (u16*)take((size_t)ROWS * CDIM * 2);                 // 8 MB
    float* cm      = (float*)take((size_t)BATCH * NCHUNK * CDIM * 4);     // 256 KB
    float* q_rt_n  = (float*)take((size_t)ROWS * 32 * 4);                 // 512 KB
    float* re_n    = (float*)take((size_t)BATCH * NCHUNK * 32 * 4);       // 8 KB

    // 1) precision staging
    cvt_bf16_kernel<<<(ROWS * CDIM + 255) / 256, 256, 0, stream>>>(x, xb, ROWS * CDIM);
    swz_weight_kernel<<<((CDIM / 2) * 3 * CDIM + 255) / 256, 256, 0, stream>>>(
        Wqkv, wqkv_swz, CDIM, 3 * CDIM);
    swz_weight_kernel<<<((CDIM / 2) * CDIM + 255) / 256, 256, 0, stream>>>(
        Wpr, wproj_swz, CDIM, CDIM);

    // 2) QKV projection (M=4096, N=3072, K=1024) with q/k/v scatter
    gemm_wmma_kernel<0><<<dim3(3 * CDIM / 64, ROWS / 128), 256, 0, stream>>>(
        xb, wqkv_swz, bqkv, ROWS, 3 * CDIM, CDIM, nullptr, q_s, k_s, v_swz);

    // 3) fused causal flash attention
    flash_attn_kernel<<<dim3(SEQ / 64, BATCH * NHEAD), 128, 0, stream>>>(
        q_s, k_s, v_swz, y_attn);

    // 4) output projection (M=4096, N=1024, K=1024) -> f32 y
    gemm_wmma_kernel<1><<<dim3(CDIM / 64, ROWS / 128), 256, 0, stream>>>(
        y_attn, wproj_swz, bpr, ROWS, CDIM, CDIM, y_out, nullptr, nullptr, nullptr);

    // 5) routing path
    chunk_mean_kernel<<<BATCH * NCHUNK, 256, 0, stream>>>(x, cm);
    router_kernel<<<BATCH * NCHUNK, 32, 0, stream>>>(cm, Wro, bro, embeds_out, re_n);
    qrt_kernel<<<ROWS, 32, 0, stream>>>(x, Wqr, bqr, q_rt_n);
    score_topk_kernel<<<ROWS, 32, 0, stream>>>(q_rt_n, re_n, scores_out, idx_out);
}
